// GAT_46205258170449
// MI455X (gfx1250) — matrix operand order repaired and checked by reading, not compiled, not run on hardware
//
#include <hip/hip_runtime.h>
#include <hip/hip_bf16.h>

typedef __attribute__((ext_vector_type(16))) __bf16 v16bf;
typedef __attribute__((ext_vector_type(8)))  float  v8f;

#define HEADS 8

union U16 { v16bf v; unsigned short u[16]; };

__device__ inline unsigned short f2bf(float f) {
  unsigned int u = __float_as_uint(f);
  unsigned int r = (u + 0x7FFFu + ((u >> 16) & 1u)) >> 16;  // RNE
  return (unsigned short)r;
}

__device__ inline void atomicMaxF(float* addr, float val) {
  if (val >= 0.0f) atomicMax((int*)addr, __float_as_int(val));
  else             atomicMin((unsigned int*)addr, __float_as_uint(val));
}

// ---------------- elementwise helpers ----------------

__global__ void k_f32_to_bf16(const float* __restrict__ a,
                              unsigned short* __restrict__ b, long long n) {
  long long i = (long long)blockIdx.x * blockDim.x + threadIdx.x;
  if (i < n) b[i] = f2bf(a[i]);
}

__global__ void k_fill(float* __restrict__ p, float v, long long n) {
  long long i = (long long)blockIdx.x * blockDim.x + threadIdx.x;
  if (i < n) p[i] = v;
}

// agg + bias -> ELU -> bf16 (input to layer-2 GEMMs)
__global__ void k_elu_bias_bf16(const float* __restrict__ agg,
                                const float* __restrict__ bias, int colmask,
                                unsigned short* __restrict__ out, long long n) {
  long long i = (long long)blockIdx.x * blockDim.x + threadIdx.x;
  if (i >= n) return;
  float v = agg[i] + bias[(int)(i & colmask)];
  v = (v > 0.0f) ? v : (__expf(v) - 1.0f);
  out[i] = f2bf(v);
}

// ---------------- weight pack: f32 W[K x Nout] -> bf16 in WMMA B lane layout ----------------
// packed index = ((ct*(K/32) + s)*32 + lane)*16 + i
// holds W[(s*32 + (lane>>4)*8 + (i&7) + (i>>3)*16)*Nout + ct*16 + (lane&15)]
// so each lane's 16 B values for one k-slab are contiguous (2 x b128 loads).

__global__ void k_pack_w(const float* __restrict__ W, unsigned short* __restrict__ Wp,
                         int K, int Nout) {
  int idx = blockIdx.x * blockDim.x + threadIdx.x;
  if (idx >= K * Nout) return;
  int i     = idx & 15;
  int lane  = (idx >> 4) & 31;
  int t     = idx >> 9;                 // ct*(K/32) + s
  int nslab = K >> 5;
  int ct    = t / nslab;
  int s     = t - ct * nslab;
  int col   = ct * 16 + (lane & 15);
  int k     = s * 32 + (lane >> 4) * 8 + (i & 7) + (i >> 3) * 16;
  Wp[idx] = f2bf(W[(size_t)k * Nout + col]);
}

// ---------------- WMMA GEMM: Y = bf16(X[MxK]) @ W + bias, K/NOUT compile-time ----------------
// One wave computes a 16x16 tile; 4 waves/block = 4 column tiles; K loop fully unrolled.

template <int K, int NOUT>
__global__ void k_gemm_bias(const unsigned short* __restrict__ X,    // M x K (bf16 bits)
                            const unsigned short* __restrict__ Wp,   // packed B
                            const float* __restrict__ bias,          // NOUT
                            float* __restrict__ Y,                   // M x NOUT
                            int M) {
  constexpr int NSLAB = K / 32;
  const int lane = threadIdx.x & 31;
  const int wave = threadIdx.x >> 5;
  const int r0 = blockIdx.x * 16;
  const int ct = blockIdx.y * 4 + wave;   // column tile, grid sized exactly
  const int c0 = ct * 16;
  const int mrow = lane & 15;
  const int half = lane >> 4;
  int rA = r0 + mrow; if (rA >= M) rA = M - 1;

  const unsigned short* xrow  = X + (size_t)rA * K;
  const unsigned short* wbase = Wp + ((size_t)ct * NSLAB * 32 + lane) * 16;

  v8f acc = {};
#pragma unroll
  for (int s = 0; s < NSLAB; ++s) {
    U16 a, b;
    // A 16x32 bf16 (ISA 7.12.2): lane row = lane&15;
    // a.u[0..7]  = K = s*32 + half*8 + {0..7}
    // a.u[8..15] = K = s*32 + half*8 + 16 + {0..7}   -> two contiguous b128 loads
    const unsigned short* xk = xrow + s * 32 + half * 8;
#pragma unroll
    for (int i = 0; i < 8; ++i) a.u[i] = xk[i];
#pragma unroll
    for (int i = 0; i < 8; ++i) a.u[8 + i] = xk[16 + i];
    // B: packed layout -> 16 contiguous shorts per lane per slab
    const unsigned short* wk = wbase + (size_t)s * (32 * 16);
#pragma unroll
    for (int i = 0; i < 16; ++i) b.u[i] = wk[i];

    acc = __builtin_amdgcn_wmma_f32_16x16x32_bf16(false, a.v, false, b.v,
                                                  (short)0, acc, false, false);
  }

  // C/D layout: vgpr v, lane -> (m = v + half*8, n = lane&15)
  const float bv = bias[c0 + mrow];
  if (r0 + 16 <= M) {                    // uniform fast path (M % 16 == 0 case)
    float* ybase = Y + ((size_t)(r0 + half * 8)) * NOUT + c0 + mrow;
#pragma unroll
    for (int v = 0; v < 8; ++v) ybase[(size_t)v * NOUT] = acc[v] + bv;
  } else {
#pragma unroll
    for (int v = 0; v < 8; ++v) {
      int rr = r0 + v + half * 8;
      if (rr < M) Y[(size_t)rr * NOUT + c0 + mrow] = acc[v] + bv;
    }
  }
}

// ---------------- edge kernels (self-loops appended virtually: e>=E -> src=dst=e-E) ----------------

template <int C>
__global__ void k_edge_logits(const int* __restrict__ srcA, const int* __restrict__ dstA,
                              int E, int Etot,
                              const float* __restrict__ xl, const float* __restrict__ xr,
                              const float* __restrict__ att,
                              float* __restrict__ logits, float* __restrict__ mbuf) {
  int gid = blockIdx.x * blockDim.x + threadIdx.x;
  if (gid >= Etot * HEADS) return;
  int h = gid & 7;
  int e = gid >> 3;
  int s = (e < E) ? srcA[e] : (e - E);
  int d = (e < E) ? dstA[e] : (e - E);
  const float* pl = xl + ((size_t)s * HEADS + h) * C;
  const float* pr = xr + ((size_t)d * HEADS + h) * C;
  const float* pa = att + h * C;
  float acc = 0.0f;
#pragma unroll
  for (int c = 0; c < C; ++c) {
    float v = pl[c] + pr[c];
    v = (v > 0.0f) ? v : 0.2f * v;           // leaky_relu(0.2)
    acc += v * pa[c];
  }
  logits[(size_t)e * HEADS + h] = acc;
  atomicMaxF(&mbuf[(size_t)d * HEADS + h], acc);
}

__global__ void k_edge_exp(const int* __restrict__ dstA, int E, int Etot,
                           float* __restrict__ logits, const float* __restrict__ mbuf,
                           float* __restrict__ sbuf) {
  int gid = blockIdx.x * blockDim.x + threadIdx.x;
  if (gid >= Etot * HEADS) return;
  int h = gid & 7;
  int e = gid >> 3;
  int d = (e < E) ? dstA[e] : (e - E);
  float p = __expf(logits[(size_t)e * HEADS + h] - mbuf[(size_t)d * HEADS + h]);
  logits[(size_t)e * HEADS + h] = p;
  atomicAdd(&sbuf[(size_t)d * HEADS + h], p);
}

template <int C>
__global__ void k_edge_scatter(const int* __restrict__ srcA, const int* __restrict__ dstA,
                               int E, int Etot,
                               const float* __restrict__ xl, const float* __restrict__ p,
                               const float* __restrict__ sbuf, float* __restrict__ out) {
  long long gid = (long long)blockIdx.x * blockDim.x + threadIdx.x;
  if (gid >= (long long)Etot * HEADS * C) return;
  int c = (int)(gid & (C - 1));
  long long t = gid / C;
  int h = (int)(t & 7);
  int e = (int)(t >> 3);
  int s = (e < E) ? srcA[e] : (e - E);
  int d = (e < E) ? dstA[e] : (e - E);
  float alpha = p[(size_t)e * HEADS + h] / (sbuf[(size_t)d * HEADS + h] + 1e-16f);
  atomicAdd(&out[((size_t)d * HEADS + h) * C + c],
            xl[((size_t)s * HEADS + h) * C + c] * alpha);
}

// ---------------- finalize: h2 = acc + bias2 ; outputs (h2, log_softmax(h2)) ----------------
// one wave (32 lanes) per node, 64 channels = 2 per lane; wave32 shfl_xor reductions

__global__ void k_finalize(float* __restrict__ out, const float* __restrict__ bias2, int N) {
  int wave = threadIdx.x >> 5;
  int lane = threadIdx.x & 31;
  int n = blockIdx.x * 4 + wave;
  if (n >= N) return;
  float v0 = out[(size_t)n * 64 + lane]      + bias2[lane];
  float v1 = out[(size_t)n * 64 + 32 + lane] + bias2[32 + lane];
  float m = fmaxf(v0, v1);
#pragma unroll
  for (int o = 16; o > 0; o >>= 1) m = fmaxf(m, __shfl_xor(m, o, 32));
  float s = __expf(v0 - m) + __expf(v1 - m);
#pragma unroll
  for (int o = 16; o > 0; o >>= 1) s += __shfl_xor(s, o, 32);
  float lse = m + __logf(s);
  out[(size_t)n * 64 + lane]      = v0;
  out[(size_t)n * 64 + 32 + lane] = v1;
  float* out2 = out + (size_t)N * 64;
  out2[(size_t)n * 64 + lane]      = v0 - lse;
  out2[(size_t)n * 64 + 32 + lane] = v1 - lse;
}

// ---------------- host launcher ----------------

static inline unsigned int nblk(long long n, int bs) {
  return (unsigned int)((n + bs - 1) / bs);
}

extern "C" void kernel_launch(void* const* d_in, const int* in_sizes, int n_in,
                              void* d_out, int out_size, void* d_ws, size_t ws_size,
                              hipStream_t stream) {
  (void)n_in; (void)out_size; (void)ws_size;
  const float* x     = (const float*)d_in[0];
  const int*   ei    = (const int*)d_in[1];
  const float* Wl1   = (const float*)d_in[2];
  const float* bl1   = (const float*)d_in[3];
  const float* Wr1   = (const float*)d_in[4];
  const float* br1   = (const float*)d_in[5];
  const float* att1  = (const float*)d_in[6];
  const float* bias1 = (const float*)d_in[7];
  const float* Wl2   = (const float*)d_in[8];
  const float* bl2   = (const float*)d_in[9];
  const float* Wr2   = (const float*)d_in[10];
  const float* br2   = (const float*)d_in[11];
  const float* att2  = (const float*)d_in[12];
  const float* bias2 = (const float*)d_in[13];

  const int DIN = 64, HC1 = 128, HC2 = 64;       // H=8, DH=16, DOUT=8
  const int N = in_sizes[0] / DIN;
  const int E = in_sizes[1] / 2;
  const int Etot = E + N;
  const int* srcA = ei;
  const int* dstA = ei + E;

  // workspace bump allocator (256B aligned)
  char* base = (char*)d_ws;
  size_t off = 0;
  auto alloc = [&](size_t bytes) -> char* {
    char* p = base + off;
    off += (bytes + 255) & ~(size_t)255;
    return p;
  };
  unsigned short* xbf   = (unsigned short*)alloc((size_t)N * DIN * 2);
  unsigned short* hbf   = (unsigned short*)alloc((size_t)N * HC1 * 2);
  unsigned short* wlbf  = (unsigned short*)alloc((size_t)8192 * 2);
  unsigned short* wrbf  = (unsigned short*)alloc((size_t)8192 * 2);
  float* xl     = (float*)alloc((size_t)N * HC1 * 4);
  float* xr     = (float*)alloc((size_t)N * HC1 * 4);
  float* agg    = (float*)alloc((size_t)N * HC1 * 4);
  float* mbuf   = (float*)alloc((size_t)N * HEADS * 4);
  float* sbuf   = (float*)alloc((size_t)N * HEADS * 4);
  float* logits = (float*)alloc((size_t)Etot * HEADS * 4);

  const int BS = 256;

  // ===== layer 1 =====
  k_f32_to_bf16<<<nblk((long long)N * DIN, BS), BS, 0, stream>>>(x, xbf, (long long)N * DIN);
  k_pack_w<<<nblk(DIN * HC1, BS), BS, 0, stream>>>(Wl1, wlbf, DIN, HC1);
  k_pack_w<<<nblk(DIN * HC1, BS), BS, 0, stream>>>(Wr1, wrbf, DIN, HC1);

  dim3 g1((N + 15) / 16, HC1 / 64);
  k_gemm_bias<64, 128><<<g1, 128, 0, stream>>>(xbf, wlbf, bl1, xl, N);
  k_gemm_bias<64, 128><<<g1, 128, 0, stream>>>(xbf, wrbf, br1, xr, N);

  k_fill<<<nblk((long long)N * HEADS, BS), BS, 0, stream>>>(mbuf, -INFINITY, (long long)N * HEADS);
  k_fill<<<nblk((long long)N * HEADS, BS), BS, 0, stream>>>(sbuf, 0.0f, (long long)N * HEADS);
  k_fill<<<nblk((long long)N * HC1, BS), BS, 0, stream>>>(agg, 0.0f, (long long)N * HC1);

  k_edge_logits<16><<<nblk((long long)Etot * HEADS, BS), BS, 0, stream>>>(
      srcA, dstA, E, Etot, xl, xr, att1, logits, mbuf);
  k_edge_exp<<<nblk((long long)Etot * HEADS, BS), BS, 0, stream>>>(
      dstA, E, Etot, logits, mbuf, sbuf);
  k_edge_scatter<16><<<nblk((long long)Etot * HEADS * 16, BS), BS, 0, stream>>>(
      srcA, dstA, E, Etot, xl, logits, sbuf, agg);

  k_elu_bias_bf16<<<nblk((long long)N * HC1, BS), BS, 0, stream>>>(
      agg, bias1, HC1 - 1, hbf, (long long)N * HC1);

  // ===== layer 2 =====
  k_pack_w<<<nblk(HC1 * HC2, BS), BS, 0, stream>>>(Wl2, wlbf, HC1, HC2);
  k_pack_w<<<nblk(HC1 * HC2, BS), BS, 0, stream>>>(Wr2, wrbf, HC1, HC2);

  dim3 g2((N + 15) / 16, HC2 / 64);
  k_gemm_bias<128, 64><<<g2, 128, 0, stream>>>(hbf, wlbf, bl2, xl, N);
  k_gemm_bias<128, 64><<<g2, 128, 0, stream>>>(hbf, wrbf, br2, xr, N);

  float* h2 = (float*)d_out;   // accumulate layer-2 output directly into first tuple slot
  k_fill<<<nblk((long long)N * HEADS, BS), BS, 0, stream>>>(mbuf, -INFINITY, (long long)N * HEADS);
  k_fill<<<nblk((long long)N * HEADS, BS), BS, 0, stream>>>(sbuf, 0.0f, (long long)N * HEADS);
  k_fill<<<nblk((long long)N * HC2, BS), BS, 0, stream>>>(h2, 0.0f, (long long)N * HC2);

  k_edge_logits<8><<<nblk((long long)Etot * HEADS, BS), BS, 0, stream>>>(
      srcA, dstA, E, Etot, xl, xr, att2, logits, mbuf);
  k_edge_exp<<<nblk((long long)Etot * HEADS, BS), BS, 0, stream>>>(
      dstA, E, Etot, logits, mbuf, sbuf);
  k_edge_scatter<8><<<nblk((long long)Etot * HEADS * 8, BS), BS, 0, stream>>>(
      srcA, dstA, E, Etot, xl, logits, sbuf, h2);

  k_finalize<<<(N + 3) / 4, 128, 0, stream>>>(h2, bias2, N);
}
